// Seq2Seq_45930380264060
// MI455X (gfx1250) — matrix operand order repaired and checked by reading, not compiled, not run on hardware
//
#include <hip/hip_runtime.h>
#include <hip/hip_bf16.h>

// Problem constants (match reference)
#define SLEN  64
#define TLEN  64
#define BATCH 32
#define EMB   1024          // == 2H
#define HID   512
#define VOC   32000
#define NLAY  4
#define GATES (4*HID)       // 2048
#define DD2   (2*HID)       // 1024
#define CAT   (10*HID)      // 5120

typedef __attribute__((ext_vector_type(16))) __bf16         v16bf;
typedef __attribute__((ext_vector_type(16))) unsigned short v16us;
typedef __attribute__((ext_vector_type(8)))  unsigned short v8us;
typedef __attribute__((ext_vector_type(8)))  float          v8f;

__device__ __forceinline__ unsigned short f2bf(float f) {
  unsigned int u = __float_as_uint(f);
  u += 0x7fffu + ((u >> 16) & 1u);          // round-to-nearest-even
  return (unsigned short)(u >> 16);
}
__device__ __forceinline__ v16bf as_bf16(v16us x) {
  union { v16us u; v16bf b; } t; t.u = x; return t.b;
}

// ---------------------------------------------------------------------------
// C[M,N](f32 or bf16) = A[M,K](bf16) * W[N,K]^T(bf16) [+bias] [+=C] [relu]
// One wave computes a 16x64 strip: one A fragment feeds 4 back-to-back
// v_wmma_f32_16x16x32_bf16 against 4 adjacent B column tiles (independent
// accumulators -> no RAW hazard, pipelines through the matrix unit).
// Inner loop is pure b128 loads + wmma (activations pre-converted to bf16).
// Strip order is column-group-major: the B weight panel is shared by
// consecutive waves and the bf16 A panel (<=21MB) stays L2-resident, so W2
// (327MB bf16) streams from HBM exactly once.
// flags: bit0 = accumulate into C, bit1 = relu, bit2 = store output as bf16
// ---------------------------------------------------------------------------
__global__ void k_gemm(const unsigned short* __restrict__ A, int lda, // [M,K] bf16
                       const unsigned short* __restrict__ W,          // [N,K] bf16
                       float* __restrict__ C, int ldc,
                       const float* __restrict__ bias,
                       int M, int N, int K, int flags)
{
  const int lane = threadIdx.x & 31;
  const int wave = threadIdx.x >> 5;
  const int ntM  = M >> 4;
  const int ngN  = N >> 6;                  // groups of 4 N-tiles
  int g = blockIdx.x * (blockDim.x >> 5) + wave;
  if (g >= ntM * ngN) return;
  const int ng   = g / ntM;                 // column-group-major ordering
  const int mt   = g - ng * ntM;
  const int half = lane >> 4;               // wave32: lanes 0-15 / 16-31
  const int l16  = lane & 15;

  // A: lane = row M; per 32-K step K in {16h..16h+15}
  const unsigned short* Ar = A + (size_t)(mt * 16 + l16) * lda + 16 * half;
  // B: lane = col N; lane-half selects K 0-15 vs 16-31; 16 contiguous bf16
  const unsigned short* Br = W + (size_t)(ng * 64 + l16) * K + 16 * half;
  const size_t bs = (size_t)16 * K;         // stride between adjacent N-tiles

  v8f acc[4];
#pragma unroll
  for (int i = 0; i < 4; ++i) acc[i] = (v8f){0.f,0.f,0.f,0.f,0.f,0.f,0.f,0.f};

  for (int kk = 0; kk < K; kk += 32) {
    union { struct { v8us lo, hi; } p; v16bf v; } a;
    a.p.lo = *(const v8us*)(Ar + kk);            // 16B aligned
    a.p.hi = *(const v8us*)(Ar + kk + 8);
    v16us b0 = *(const v16us*)(Br + kk);         // 32B aligned
    v16us b1 = *(const v16us*)(Br + bs + kk);
    v16us b2 = *(const v16us*)(Br + 2 * bs + kk);
    v16us b3 = *(const v16us*)(Br + 3 * bs + kk);
    if (kk + 32 < K) __builtin_prefetch(Br + kk + 32, 0, 1);   // global_prefetch_b8
    acc[0] = __builtin_amdgcn_wmma_f32_16x16x32_bf16(false, a.v, false, as_bf16(b0),
                                                     (short)0, acc[0], false, false);
    acc[1] = __builtin_amdgcn_wmma_f32_16x16x32_bf16(false, a.v, false, as_bf16(b1),
                                                     (short)0, acc[1], false, false);
    acc[2] = __builtin_amdgcn_wmma_f32_16x16x32_bf16(false, a.v, false, as_bf16(b2),
                                                     (short)0, acc[2], false, false);
    acc[3] = __builtin_amdgcn_wmma_f32_16x16x32_bf16(false, a.v, false, as_bf16(b3),
                                                     (short)0, acc[3], false, false);
  }

#pragma unroll
  for (int i = 0; i < 4; ++i) {
    const int   col = ng * 64 + i * 16 + l16;
    const float bv  = bias ? bias[col] : 0.0f;
#pragma unroll
    for (int v = 0; v < 8; ++v) {            // C/D: M = 8*half + v, N = l16
      int    row = mt * 16 + half * 8 + v;
      size_t o   = (size_t)row * ldc + col;
      float  val = acc[i][v] + bv;
      if (flags & 1) val += C[o];
      if (flags & 2) val = fmaxf(val, 0.0f);
      if (flags & 4) ((unsigned short*)C)[o] = f2bf(val);
      else           C[o] = val;
    }
  }
}

// ---------------------------------------------------------------------------
__global__ void k_cvt(unsigned short* __restrict__ dst, const float* __restrict__ src,
                      long long n)
{
  long long i = (long long)blockIdx.x * blockDim.x + threadIdx.x;
  if (i < n) dst[i] = f2bf(src[i]);
}

__global__ void k_zero(float* p, int n)
{
  int i = blockIdx.x * blockDim.x + threadIdx.x;
  if (i < n) p[i] = 0.f;
}

// out_bf16[(t*B+b)*E + e] = emb[tok[(shift ? max(t-1,0) : t)*B + b]*E + e]
__global__ void k_embed(unsigned short* __restrict__ out, const float* __restrict__ emb,
                        const int* __restrict__ tok, int rows, int shift)
{
  long long i = (long long)blockIdx.x * blockDim.x + threadIdx.x;
  if (i >= (long long)rows * EMB) return;
  int r = (int)(i / EMB), e = (int)(i % EMB);
  int t = r / BATCH, b = r - t * BATCH;
  int tt = (shift && t > 0) ? (t - 1) : t;
  out[i] = f2bf(emb[(size_t)tok[tt * BATCH + b] * EMB + e]);
}

// gates[B,4H] (PyTorch order i,f,g,o) -> update h (f32 + bf16), c;
// optionally scatter h into a concat buffer (f32 and bf16 views)
__global__ void k_lstm(const float* __restrict__ gates,
                       float* __restrict__ h, unsigned short* __restrict__ hbf,
                       float* __restrict__ c,
                       float* __restrict__ xout, unsigned short* __restrict__ xoutbf,
                       int xld, int xoff)
{
  int i = blockIdx.x * blockDim.x + threadIdx.x;
  if (i >= BATCH * HID) return;
  int b = i / HID, j = i - b * HID;
  const float* g = gates + (size_t)b * GATES;
  float gi = g[j], gf = g[HID + j], gg = g[2 * HID + j], go = g[3 * HID + j];
  float si = 1.f / (1.f + __expf(-gi));
  float sf = 1.f / (1.f + __expf(-gf));
  float so = 1.f / (1.f + __expf(-go));
  float c2 = sf * c[i] + si * tanhf(gg);
  float h2 = so * tanhf(c2);
  c[i] = c2; h[i] = h2; hbf[i] = f2bf(h2);
  size_t xo = (size_t)b * xld + xoff + j;
  if (xout)   xout[xo]   = h2;
  if (xoutbf) xoutbf[xo] = f2bf(h2);
}

// dot-product attention: one block per batch element
__global__ void k_attention(const float* __restrict__ xin,  // [B, 2H]
                            const float* __restrict__ oe,   // [S*B, 2H] (seq-major)
                            float* __restrict__ w)          // [B, 2H]
{
  int b = blockIdx.x;
  __shared__ float a[SLEN];
  if (threadIdx.x < SLEN) {
    const float* x = xin + (size_t)b * DD2;
    const float* o = oe + ((size_t)threadIdx.x * BATCH + b) * DD2;
    float acc = 0.f;
    for (int j = 0; j < DD2; ++j) acc += x[j] * o[j];
    a[threadIdx.x] = acc;
  }
  __syncthreads();
  if (threadIdx.x == 0) {
    float m = a[0];
    for (int s = 1; s < SLEN; ++s) m = fmaxf(m, a[s]);
    float sum = 0.f;
    for (int s = 0; s < SLEN; ++s) { float v = __expf(a[s] - m); a[s] = v; sum += v; }
    float inv = 1.f / sum;
    for (int s = 0; s < SLEN; ++s) a[s] *= inv;
  }
  __syncthreads();
  for (int j = threadIdx.x; j < DD2; j += blockDim.x) {
    float acc = 0.f;
    for (int s = 0; s < SLEN; ++s) acc += a[s] * oe[((size_t)s * BATCH + b) * DD2 + j];
    w[(size_t)b * DD2 + j] = acc;
  }
}

// xcat_bf16[b, :] = [ h2.transpose(2,0,1,3).reshape(B,8H) , attn_w(b, 2H) ]
__global__ void k_xcat(unsigned short* __restrict__ dst,    // [B, 10H] row block (bf16)
                       const float* __restrict__ hstate,    // [L*2*B*H]
                       const float* __restrict__ w)         // [B, 2H]
{
  int i = blockIdx.x * blockDim.x + threadIdx.x;
  if (i >= BATCH * CAT) return;
  int b = i / CAT, j = i - b * CAT;
  float v;
  if (j < 2 * NLAY * HID) {
    int l = j / DD2; int rem = j - l * DD2; int d = rem / HID; int hh = rem - d * HID;
    v = hstate[(((size_t)l * 2 + d) * BATCH + b) * HID + hh];
  } else {
    v = w[(size_t)b * DD2 + (j - 2 * NLAY * HID)];
  }
  dst[(size_t)b * CAT + j] = f2bf(v);
}

// log_softmax over V; row = t*B+b from [T*B,V] scratch -> out[b,t,v]
__global__ void k_logsoftmax(const float* __restrict__ logits, float* __restrict__ out)
{
  int row = blockIdx.x;
  int t = row / BATCH, b = row - t * BATCH;
  const float* x = logits + (size_t)row * VOC;
  __shared__ float red[256];
  float m = -3.4e38f;
  for (int v = threadIdx.x; v < VOC; v += 256) m = fmaxf(m, x[v]);
  red[threadIdx.x] = m; __syncthreads();
  for (int s = 128; s > 0; s >>= 1) {
    if (threadIdx.x < s) red[threadIdx.x] = fmaxf(red[threadIdx.x], red[threadIdx.x + s]);
    __syncthreads();
  }
  m = red[0]; __syncthreads();
  float sum = 0.f;
  for (int v = threadIdx.x; v < VOC; v += 256) sum += __expf(x[v] - m);
  red[threadIdx.x] = sum; __syncthreads();
  for (int s = 128; s > 0; s >>= 1) {
    if (threadIdx.x < s) red[threadIdx.x] += red[threadIdx.x + s];
    __syncthreads();
  }
  float lse = m + __logf(red[0]);
  float* o = out + ((size_t)b * TLEN + t) * VOC;
  for (int v = threadIdx.x; v < VOC; v += 256) o[v] = x[v] - lse;
}

// ---------------------------------------------------------------------------
extern "C" void kernel_launch(void* const* d_in, const int* in_sizes, int n_in,
                              void* d_out, int out_size, void* d_ws, size_t ws_size,
                              hipStream_t stream)
{
  (void)in_sizes; (void)n_in; (void)out_size; (void)ws_size;
  const int*   src  = (const int*)  d_in[0];
  const int*   trg  = (const int*)  d_in[1];
  const float* embS = (const float*)d_in[4];
  const float* embT = (const float*)d_in[5];
  const float* eWih = (const float*)d_in[6];
  const float* eWhh = (const float*)d_in[7];
  const float* eB   = (const float*)d_in[8];
  const float* dWih = (const float*)d_in[9];
  const float* dWhh = (const float*)d_in[10];
  const float* dB   = (const float*)d_in[11];
  const float* W1   = (const float*)d_in[12];
  const float* b1   = (const float*)d_in[13];
  const float* W2   = (const float*)d_in[14];
  const float* b2   = (const float*)d_in[15];
  float* out = (float*)d_out;

  // workspace carve-out (256B aligned)
  char* base = (char*)d_ws;
  size_t off = 0;
  auto alloc = [&](size_t bytes) -> char* {
    char* p = base + off; off = (off + bytes + 255) & ~(size_t)255; return p;
  };
  unsigned short* eWih_bf = (unsigned short*)alloc((size_t)NLAY*2*GATES*EMB*2);
  unsigned short* eWhh_bf = (unsigned short*)alloc((size_t)NLAY*2*GATES*HID*2);
  unsigned short* dWih_bf = (unsigned short*)alloc((size_t)NLAY*2*GATES*EMB*2);
  unsigned short* dWhh_bf = (unsigned short*)alloc((size_t)NLAY*2*GATES*HID*2);
  unsigned short* W1_bf   = (unsigned short*)alloc((size_t)CAT*CAT*2);
  unsigned short* W2_bf   = (unsigned short*)alloc((size_t)VOC*CAT*2);
  unsigned short* xs      = (unsigned short*)alloc((size_t)SLEN*BATCH*EMB*2);   // bf16
  float*          xbufA   = (float*)alloc((size_t)SLEN*BATCH*DD2*4);
  float*          xbufB   = (float*)alloc((size_t)SLEN*BATCH*DD2*4);
  unsigned short* xbufAbf = (unsigned short*)alloc((size_t)SLEN*BATCH*DD2*2);
  unsigned short* xbufBbf = (unsigned short*)alloc((size_t)SLEN*BATCH*DD2*2);
  float*          XpreF   = (float*)alloc((size_t)SLEN*BATCH*GATES*4);
  float*          XpreB   = (float*)alloc((size_t)SLEN*BATCH*GATES*4);
  float*          hstate  = (float*)alloc((size_t)NLAY*2*BATCH*HID*4);
  unsigned short* hbf     = (unsigned short*)alloc((size_t)NLAY*2*BATCH*HID*2);
  float*          cstate  = (float*)alloc((size_t)NLAY*2*BATCH*HID*4);
  unsigned short* decin   = (unsigned short*)alloc((size_t)TLEN*BATCH*EMB*2);   // bf16
  float*          dXpre0  = (float*)alloc((size_t)2*TLEN*BATCH*GATES*4);
  float*          gbuf    = (float*)alloc((size_t)2*BATCH*GATES*4);
  float*          xinA    = (float*)alloc((size_t)BATCH*DD2*4);
  float*          xinB    = (float*)alloc((size_t)BATCH*DD2*4);
  unsigned short* xinAbf  = (unsigned short*)alloc((size_t)BATCH*DD2*2);
  unsigned short* xinBbf  = (unsigned short*)alloc((size_t)BATCH*DD2*2);
  float*          wbuf    = (float*)alloc((size_t)BATCH*DD2*4);
  unsigned short* Xcat    = (unsigned short*)alloc((size_t)TLEN*BATCH*CAT*2);   // bf16
  unsigned short* hidb    = (unsigned short*)alloc((size_t)TLEN*BATCH*CAT*2);   // bf16
  float*          logits  = (float*)alloc((size_t)TLEN*BATCH*VOC*4);

  auto cvt = [&](unsigned short* d, const float* s, long long n) {
    k_cvt<<<(unsigned)((n + 255) / 256), 256, 0, stream>>>(d, s, n);
  };
  auto gemm = [&](const unsigned short* A, int lda, const unsigned short* Wb,
                  float* C, int ldc, const float* bias, int M, int N, int K, int flags) {
    int groups = (M / 16) * (N / 64);
    k_gemm<<<(groups + 7) / 8, 256, 0, stream>>>(A, lda, Wb, C, ldc, bias, M, N, K, flags);
  };
  auto zero = [&](void* p, size_t bytes) {     // bytes must be multiple of 4
    k_zero<<<((int)(bytes / 4) + 255) / 256, 256, 0, stream>>>((float*)p, (int)(bytes / 4));
  };
  auto lstm = [&](const float* g, float* h, unsigned short* hb, float* c,
                  float* xo, unsigned short* xob, int xld, int xoff) {
    k_lstm<<<(BATCH * HID + 255) / 256, 256, 0, stream>>>(g, h, hb, c, xo, xob, xld, xoff);
  };

  // 0) weights -> bf16 (read once per call; deterministic)
  cvt(eWih_bf, eWih, (long long)NLAY*2*GATES*EMB);
  cvt(eWhh_bf, eWhh, (long long)NLAY*2*GATES*HID);
  cvt(dWih_bf, dWih, (long long)NLAY*2*GATES*EMB);
  cvt(dWhh_bf, dWhh, (long long)NLAY*2*GATES*HID);
  cvt(W1_bf,   W1,   (long long)CAT*CAT);
  cvt(W2_bf,   W2,   (long long)VOC*CAT);

  // 1) embeddings straight to bf16 (decoder shift: step0->trg[0], step i->trg[i-1])
  k_embed<<<((long long)SLEN*BATCH*EMB + 255) / 256, 256, 0, stream>>>(xs, embS, src, SLEN*BATCH, 0);
  k_embed<<<((long long)TLEN*BATCH*EMB + 255) / 256, 256, 0, stream>>>(decin, embT, trg, TLEN*BATCH, 1);

  // 2) encoder: batched x@Wih^T + b per (layer,dir), then sequential h@Whh^T
  const unsigned short* curbf = xs;
  float*          nxt   = xbufA;
  unsigned short* nxtbf = xbufAbf;
  float*          oeF   = nullptr;
  for (int l = 0; l < NLAY; ++l) {
    for (int dir = 0; dir < 2; ++dir) {
      gemm(curbf, EMB, eWih_bf + (size_t)(l*2+dir)*GATES*EMB,
           dir ? XpreB : XpreF, GATES, eB + (size_t)(l*2+dir)*GATES,
           SLEN*BATCH, GATES, EMB, 0);
    }
    for (int dir = 0; dir < 2; ++dir) {
      float*          h  = hstate + (size_t)(l*2+dir)*BATCH*HID;
      unsigned short* hb = hbf    + (size_t)(l*2+dir)*BATCH*HID;
      float*          c  = cstate + (size_t)(l*2+dir)*BATCH*HID;
      zero(h, (size_t)BATCH*HID*4); zero(c, (size_t)BATCH*HID*4);
      zero(hb, (size_t)BATCH*HID*2);
      const unsigned short* Wh = eWhh_bf + (size_t)(l*2+dir)*GATES*HID;
      float* Xp = dir ? XpreB : XpreF;
      for (int s = 0; s < SLEN; ++s) {
        int t = dir ? (SLEN - 1 - s) : s;
        float* g = Xp + (size_t)t*BATCH*GATES;
        gemm(hb, HID, Wh, g, GATES, nullptr, BATCH, GATES, HID, 1);  // gates += h@Whh^T
        lstm(g, h, hb, c,
             nxt + (size_t)t*BATCH*DD2, nxtbf + (size_t)t*BATCH*DD2, DD2, dir*HID);
      }
    }
    curbf = nxtbf; oeF = nxt;
    if (nxt == xbufA) { nxt = xbufB; nxtbf = xbufBbf; }
    else              { nxt = xbufA; nxtbf = xbufAbf; }
  }
  const float* oe = oeF;   // encoder output [S*B, 2H] f32; final h/c in hstate/hbf/cstate

  // 3) decoder layer-0 input GEMMs batched over all T
  for (int dir = 0; dir < 2; ++dir) {
    gemm(decin, EMB, dWih_bf + (size_t)dir*GATES*EMB,
         dXpre0 + (size_t)dir*TLEN*BATCH*GATES, GATES,
         dB + (size_t)dir*GATES, TLEN*BATCH, GATES, EMB, 0);
  }

  // 4) decoder time loop (recurrence + attention + xcat)
  for (int t = 0; t < TLEN; ++t) {
    for (int l = 0; l < NLAY; ++l) {
      const unsigned short* linbf = (l == 0) ? nullptr : ((l & 1) ? xinAbf : xinBbf);
      float*          lout   = (l & 1) ? xinB   : xinA;
      unsigned short* loutbf = (l & 1) ? xinBbf : xinAbf;
      for (int dir = 0; dir < 2; ++dir) {
        float* g;
        if (l == 0) {
          g = dXpre0 + (size_t)dir*TLEN*BATCH*GATES + (size_t)t*BATCH*GATES;
        } else {
          g = gbuf + (size_t)dir*BATCH*GATES;
          gemm(linbf, DD2, dWih_bf + (size_t)(l*2+dir)*GATES*EMB, g, GATES,
               dB + (size_t)(l*2+dir)*GATES, BATCH, GATES, DD2, 0);
        }
        float*          h  = hstate + (size_t)(l*2+dir)*BATCH*HID;
        unsigned short* hb = hbf    + (size_t)(l*2+dir)*BATCH*HID;
        float*          c  = cstate + (size_t)(l*2+dir)*BATCH*HID;
        gemm(hb, HID, dWhh_bf + (size_t)(l*2+dir)*GATES*HID, g, GATES,
             nullptr, BATCH, GATES, HID, 1);
        lstm(g, h, hb, c, lout, loutbf, DD2, dir*HID);
      }
    }
    k_attention<<<BATCH, 256, 0, stream>>>(xinB, oe, wbuf);     // layer-3 out = xinB
    k_xcat<<<(BATCH*CAT + 255) / 256, 256, 0, stream>>>(
        Xcat + (size_t)t*BATCH*CAT, hstate, wbuf);
  }

  // 5) classifier batched over all T*B rows (the FLOP-dominant GEMMs)
  gemm(Xcat, CAT, W1_bf, (float*)hidb, CAT, b1, TLEN*BATCH, CAT, CAT, 2 | 4); // relu, bf16 out
  gemm(hidb, CAT, W2_bf, logits, VOC, b2, TLEN*BATCH, VOC, CAT, 0);
  k_logsoftmax<<<TLEN*BATCH, 256, 0, stream>>>(logits, out);
}
